// TPAttention_22780506538510
// MI455X (gfx1250) — matrix operand order repaired and checked by reading, not compiled, not run on hardware
//
#include <hip/hip_runtime.h>

// ---------------- problem constants (from reference) ----------------
constexpr int B_  = 2;
constexpr int T_  = 2048;
constexpr int D_  = 1024;
constexpr int NHh = 16;
constexpr int DHh = 64;
constexpr int CHN = 80;          // NH + DH
constexpr int RQn = 6;
constexpr int RKn = 2;
constexpr int TOK = B_ * T_;     // 4096

typedef __attribute__((ext_vector_type(16))) __bf16 v16bf;
typedef __attribute__((ext_vector_type(8)))  float  v8f;
typedef __attribute__((ext_vector_type(4)))  unsigned int v4u;
typedef __attribute__((ext_vector_type(8)))  int v8i;
typedef __attribute__((ext_vector_type(4)))  int v4i;

__device__ __forceinline__ v8f wmma_bf16(v16bf a, v16bf b, v8f c) {
  return __builtin_amdgcn_wmma_f32_16x16x32_bf16(false, a, false, b, (short)0, c, false, false);
}

// A-matrix 16x32 (16-bit) fragment K index for pair p (0..7), lane-half h
__device__ __forceinline__ int a_frag_k(int p, int h) {
  return (p < 4) ? (2 * p + 8 * h) : (16 + 2 * (p - 4) + 8 * h);
}
// B-matrix 32x16 (16-bit) fragment K index for pair p, lane-half h
__device__ __forceinline__ int b_frag_k(int p, int h) {
  return (p < 4) ? (2 * p + 8 * h) : (16 + 2 * (p - 4) + 8 * h);
}

// ---------------- Tensor Data Mover: 2D tile (global -> LDS) ----------------
// Builds a D# per CDNA5 ISA and issues
// TENSOR_LOAD_TO_LDS. data_size = 2 bytes (bf16). Tracked by TENSORcnt.
#if defined(__has_builtin)
#if __has_builtin(__builtin_amdgcn_tensor_load_to_lds)
#define HAVE_TDM 1
#endif
#endif
#ifndef HAVE_TDM
#define HAVE_TDM 0
#endif

#if HAVE_TDM
__device__ __forceinline__ void tdm_load_2d_bf16(unsigned lds_addr,
                                                 const void* gaddr,
                                                 unsigned tile_w, unsigned tile_h,
                                                 unsigned avail_w, unsigned avail_h,
                                                 unsigned row_stride_elems) {
  unsigned long long ga = (unsigned long long)(size_t)gaddr;
  v4u g0;
  g0[0] = 1u;                                            // count=1 valid user D#
  g0[1] = lds_addr;                                      // lds_addr[31:0]
  g0[2] = (unsigned)(ga & 0xffffffffu);                  // global_addr[31:0]
  g0[3] = (unsigned)((ga >> 32) & 0x01ffffffu)           // global_addr[56:32]
          | (2u << 30);                                  // type = 2 ("image")
  v8i g1;
  g1[0] = (int)(1u << 16);                               // data_size=1 -> 2 bytes
  g1[1] = (int)((avail_w & 0xffffu) << 16);              // tensor_dim0[15:0] @ [63:48]
  g1[2] = (int)(((avail_w >> 16) & 0xffffu)              // tensor_dim0[31:16]
                | ((avail_h & 0xffffu) << 16));          // tensor_dim1[15:0]
  g1[3] = (int)(((avail_h >> 16) & 0xffffu)              // tensor_dim1[31:16]
                | ((tile_w & 0xffffu) << 16));           // tile_dim0 @ [127:112]
  g1[4] = (int)(tile_h & 0xffffu);                       // tile_dim1 @ [143:128]
  g1[5] = (int)row_stride_elems;                         // tensor_dim0_stride[31:0]
  g1[6] = 0;                                             // stride[47:32]=0, dim1_stride lo=0
  g1[7] = 0;
  v4i z4 = {0, 0, 0, 0};
#if defined(__clang_major__) && (__clang_major__ >= 23)
  v8i z8 = {0, 0, 0, 0, 0, 0, 0, 0};
  __builtin_amdgcn_tensor_load_to_lds(g0, g1, z4, z4, z8, 0);
#else
  __builtin_amdgcn_tensor_load_to_lds(g0, g1, z4, z4, 0);
#endif
}
#endif

// ---------------- fp32 -> bf16 conversion ----------------
__global__ void f32_to_bf16_kernel(const float* __restrict__ in,
                                   __bf16* __restrict__ out, int n) {
  int i = blockIdx.x * blockDim.x + threadIdx.x;
  if (i < n) out[i] = (__bf16)in[i];
}

// ---------------- GEMM: C(M,N) f32 = A(M,K) bf16 @ B(N,K)^T bf16 ----------------
// block: 256 threads (8 waves); tile 64(M) x 128(N); K-step 32.
// Tiles staged into LDS by the Tensor Data Mover (wave 0 issues two D#s,
// waits s_wait_tensorcnt 0, then the workgroup barrier publishes them).
__global__ __launch_bounds__(256) void gemm_bf16_nt(const __bf16* __restrict__ A,
                                                    const __bf16* __restrict__ Bm,
                                                    float* __restrict__ C,
                                                    int M, int N, int K) {
  __shared__ __attribute__((aligned(16))) __bf16 As[64][32];
  __shared__ __attribute__((aligned(16))) __bf16 Bs[128][32];

  const int tid  = threadIdx.x;
  const int lane = tid & 31;
  const int wid  = tid >> 5;
  const int m0   = blockIdx.y * 64;
  const int n0   = blockIdx.x * 128;
  const int mrow0 = (wid & 3) * 16;   // wave's M sub-tile (16 rows)
  const int nbase = (wid >> 2) * 64;  // wave's N sub-tile (four 16-wide groups)

  v8f acc[4];
#pragma unroll
  for (int s = 0; s < 4; ++s) acc[s] = v8f{0.f,0.f,0.f,0.f,0.f,0.f,0.f,0.f};

  const unsigned availB = (N > n0) ? (unsigned)(N - n0) : 0u;

  for (int k0 = 0; k0 < K; k0 += 32) {
#if HAVE_TDM
    if (wid == 0) {
      tdm_load_2d_bf16((unsigned)(size_t)&As[0][0],
                       A + (size_t)m0 * K + k0, 32u, 64u, 32u, 64u, (unsigned)K);
      tdm_load_2d_bf16((unsigned)(size_t)&Bs[0][0],
                       Bm + (size_t)n0 * K + k0, 32u, 128u, 32u,
                       (availB < 128u) ? availB : 128u, (unsigned)K);
      __builtin_amdgcn_s_wait_tensorcnt(0);
    }
    if (k0 + 32 < K)  // warm next A K-slice in GL2 (global_prefetch_b8)
      __builtin_prefetch(A + (size_t)(m0 + (tid & 63)) * K + k0 + 32, 0, 1);
#else
    // fallback: cooperative staging
    {
      int lrow = tid >> 2, lcol = (tid & 3) * 8;
      *(uint4*)&As[lrow][lcol] =
          *(const uint4*)(A + (size_t)(m0 + lrow) * K + k0 + lcol);
      int r0 = tid >> 1, c0 = (tid & 1) * 16;
#pragma unroll
      for (int rep = 0; rep < 2; ++rep) {
        int n = n0 + r0 + rep * 0;  // two 8-elem chunks per thread
        int cc = c0 + rep * 8;
        if (n < N)
          *(uint4*)&Bs[r0][cc] = *(const uint4*)(Bm + (size_t)n * K + k0 + cc);
        else {
          uint4 z = {0u,0u,0u,0u};
          *(uint4*)&Bs[r0][cc] = z;
        }
      }
    }
#endif
    __syncthreads();

    const int half = lane >> 4;
    const int arow = lane & 15;
    v16bf af;
#pragma unroll
    for (int p = 0; p < 8; ++p) {
      int ka = a_frag_k(p, half);
      af[2*p]   = As[mrow0 + arow][ka];
      af[2*p+1] = As[mrow0 + arow][ka + 1];
    }
#pragma unroll
    for (int s = 0; s < 4; ++s) {
      v16bf bf;
#pragma unroll
      for (int p = 0; p < 8; ++p) {
        int kb = b_frag_k(p, half);
        bf[2*p]   = Bs[nbase + s * 16 + arow][kb];
        bf[2*p+1] = Bs[nbase + s * 16 + arow][kb + 1];
      }
      acc[s] = wmma_bf16(af, bf, acc[s]);
    }
    __syncthreads();
  }

  const int cc = lane & 15;
#pragma unroll
  for (int i = 0; i < 8; ++i) {
    int cr = (lane < 16) ? i : (i + 8);
    int gm = m0 + mrow0 + cr;
#pragma unroll
    for (int s = 0; s < 4; ++s) {
      int gn = n0 + nbase + s * 16 + cc;
      if (gn < N) C[(size_t)gm * N + gn] = acc[s][i];
    }
  }
}

// ---------------- q/k/v build: rank contraction + RoPE, bf16 out [B,H,T,D] ----------------
__global__ void qkv_build_kernel(const float* __restrict__ abq,
                                 const float* __restrict__ abkv,
                                 __bf16* __restrict__ q,
                                 __bf16* __restrict__ k,
                                 __bf16* __restrict__ v) {
  int idx = blockIdx.x * blockDim.x + threadIdx.x;
  if (idx >= TOK * NHh * DHh) return;
  int d   = idx & 63;
  int h   = (idx >> 6) & 15;
  int tok = idx >> 10;
  int t = tok & (T_ - 1);
  int b = tok >> 11;

  // RoPE angle: inv_freq = 10000^{-(d%32)/32}
  float fi  = (float)(d & 31) * (1.0f / 32.0f);
  float ang = (float)t * __expf(-fi * 9.210340371976184f);
  float cs, sn;
  __sincosf(ang, &sn, &cs);

  const float* aq = abq + (size_t)tok * (RQn * CHN);
  float qv = 0.f;
#pragma unroll
  for (int r = 0; r < RQn; ++r)
    qv += aq[r * CHN + h] * aq[r * CHN + NHh + d];
  qv *= (1.0f / RQn);

  const float* akv = abkv + (size_t)tok * (4 * CHN);
  float kv = 0.f, vv = 0.f;
#pragma unroll
  for (int r = 0; r < 4; ++r) {
    const float* brow = akv + r * CHN + NHh;
    float br = (d < 32) ? (brow[d] * cs + brow[d + 32] * sn)
                        : (-brow[d - 32] * sn + brow[d] * cs);
    float contrib = akv[r * CHN + h] * br;
    if (r < RKn) kv += contrib; else vv += contrib;
  }
  kv *= 0.5f;  // / RK
  vv *= 0.5f;  // / RV

  size_t o = ((size_t)(b * NHh + h) * T_ + t) * DHh + d;
  q[o] = (__bf16)qv;
  k[o] = (__bf16)kv;
  v[o] = (__bf16)vv;
}

// ---------------- flash attention: 1 wave per (b,h, 16-query tile) ----------------
__global__ __launch_bounds__(32) void attn_kernel(const __bf16* __restrict__ Q,
                                                  const __bf16* __restrict__ K,
                                                  const __bf16* __restrict__ V,
                                                  __bf16* __restrict__ Yatt) {
  const int bh  = blockIdx.x;           // b*NH + h
  const int qt0 = blockIdx.y * 16;
  const int b   = bh >> 4;
  const int h   = bh & 15;
  const __bf16* Qb = Q + (size_t)bh * T_ * DHh;
  const __bf16* Kb = K + (size_t)bh * T_ * DHh;
  const __bf16* Vb = V + (size_t)bh * T_ * DHh;

  const int lane = threadIdx.x & 31;
  const int half = lane >> 4;
  const int l15  = lane & 15;

  // Q fragments: two K-steps over d (0..31, 32..63)
  v16bf qf[2];
  {
    const __bf16* qrow = Qb + (size_t)(qt0 + l15) * DHh;
#pragma unroll
    for (int ks = 0; ks < 2; ++ks)
#pragma unroll
      for (int p = 0; p < 8; ++p) {
        int dd = ks * 32 + a_frag_k(p, half);
        qf[ks][2*p]   = qrow[dd];
        qf[ks][2*p+1] = qrow[dd + 1];
      }
  }

  v8f o[4];
  float mrun[8], lrun[8];
#pragma unroll
  for (int c = 0; c < 4; ++c) o[c] = v8f{0.f,0.f,0.f,0.f,0.f,0.f,0.f,0.f};
#pragma unroll
  for (int i = 0; i < 8; ++i) { mrun[i] = -3.0e38f; lrun[i] = 0.f; }

  __shared__ __attribute__((aligned(16))) __bf16 Pl[16][32];
  const float scale = 0.125f;  // 1/sqrt(64)
  const int kend = qt0 + 16;

  for (int kb = 0; kb < kend; kb += 32) {
    // ---- S = Q K^T for 16q x 32k ----
    v8f s0 = v8f{0.f,0.f,0.f,0.f,0.f,0.f,0.f,0.f};
    v8f s1 = s0;
#pragma unroll
    for (int ks = 0; ks < 2; ++ks) {
      v16bf kf0, kf1;
      const __bf16* k0p = Kb + (size_t)(kb + l15) * DHh;       // key n
      const __bf16* k1p = Kb + (size_t)(kb + 16 + l15) * DHh;  // key n+16
#pragma unroll
      for (int p = 0; p < 8; ++p) {
        int dd = ks * 32 + b_frag_k(p, half);
        kf0[2*p]   = k0p[dd];
        kf0[2*p+1] = k0p[dd + 1];
        kf1[2*p]   = k1p[dd];
        kf1[2*p+1] = k1p[dd + 1];
      }
      s0 = wmma_bf16(qf[ks], kf0, s0);
      s1 = wmma_bf16(qf[ks], kf1, s1);
    }

    // ---- online softmax (rows live per-VGPR within 16-lane halves) ----
#pragma unroll
    for (int i = 0; i < 8; ++i) {
      int qrow = qt0 + ((lane < 16) ? i : (i + 8));
      int key0 = kb + l15;
      int key1 = key0 + 16;
      float e0 = s0[i] * scale;
      float e1 = s1[i] * scale;
      if (key0 > qrow) e0 = -3.0e38f;
      if (key1 > qrow) e1 = -3.0e38f;
      float mx = fmaxf(e0, e1);
#pragma unroll
      for (int msk = 1; msk < 16; msk <<= 1)
        mx = fmaxf(mx, __shfl_xor(mx, msk, 32));
      float mnew = fmaxf(mrun[i], mx);
      float sc = __expf(mrun[i] - mnew);
      float p0 = __expf(e0 - mnew);
      float p1 = __expf(e1 - mnew);
      float rs = p0 + p1;
#pragma unroll
      for (int msk = 1; msk < 16; msk <<= 1)
        rs += __shfl_xor(rs, msk, 32);
      lrun[i] = lrun[i] * sc + rs;
      mrun[i] = mnew;
      o[0][i] *= sc; o[1][i] *= sc; o[2][i] *= sc; o[3][i] *= sc;
      int pr = (lane < 16) ? i : (i + 8);
      Pl[pr][l15]      = (__bf16)p0;
      Pl[pr][l15 + 16] = (__bf16)p1;
    }
    __syncthreads();

    // ---- re-shape P into A-fragment layout ----
    v16bf pf;
#pragma unroll
    for (int p = 0; p < 8; ++p) {
      int kk = a_frag_k(p, half);
      pf[2*p]   = Pl[l15][kk];
      pf[2*p+1] = Pl[l15][kk + 1];
    }

    // ---- O += P V : 4 d-chunks of 16 ----
#pragma unroll
    for (int ch = 0; ch < 4; ++ch) {
      v16bf vf;
#pragma unroll
      for (int p = 0; p < 8; ++p) {
        int key = kb + b_frag_k(p, half);
        vf[2*p]   = Vb[(size_t)key * DHh + ch * 16 + l15];
        vf[2*p+1] = Vb[(size_t)(key + 1) * DHh + ch * 16 + l15];
      }
      o[ch] = wmma_bf16(pf, vf, o[ch]);
    }
    __syncthreads();
  }

  // ---- normalize and store [B,T,H*D] bf16 ----
#pragma unroll
  for (int i = 0; i < 8; ++i) {
    float inv = 1.0f / lrun[i];
    int t = qt0 + ((lane < 16) ? i : (i + 8));
    size_t base = ((size_t)(b * T_ + t)) * D_ + h * DHh;
#pragma unroll
    for (int ch = 0; ch < 4; ++ch)
      Yatt[base + ch * 16 + l15] = (__bf16)(o[ch][i] * inv);
  }
}

// ---------------- host launcher ----------------
extern "C" void kernel_launch(void* const* d_in, const int* in_sizes, int n_in,
                              void* d_out, int out_size, void* d_ws, size_t ws_size,
                              hipStream_t stream) {
  (void)in_sizes; (void)n_in; (void)out_size; (void)ws_size;
  const float* x    = (const float*)d_in[0];   // [B,T,D]
  const float* Wabq = (const float*)d_in[1];   // [480,1024]
  const float* Wkv  = (const float*)d_in[2];   // [320,1024]
  const float* Wo   = (const float*)d_in[3];   // [1024,1024]
  float* out = (float*)d_out;                  // [B,T,D] f32

  char* ws = (char*)d_ws;
  size_t off = 0;
  auto alloc = [&](size_t bytes) -> void* {
    void* p = ws + off;
    off = (off + bytes + 255) & ~(size_t)255;
    return p;
  };

  const int NQ  = RQn * CHN;        // 480
  const int NKV = 4 * CHN;          // 320
  __bf16* xb    = (__bf16*)alloc((size_t)TOK * D_ * 2);
  __bf16* Wq_b  = (__bf16*)alloc((size_t)NQ * D_ * 2);
  __bf16* Wkv_b = (__bf16*)alloc((size_t)NKV * D_ * 2);
  __bf16* Wo_b  = (__bf16*)alloc((size_t)D_ * D_ * 2);
  float*  abq   = (float*) alloc((size_t)TOK * NQ * 4);
  float*  abkv  = (float*) alloc((size_t)TOK * NKV * 4);
  __bf16* q_l   = (__bf16*)alloc((size_t)TOK * D_ * 2);
  __bf16* k_l   = (__bf16*)alloc((size_t)TOK * D_ * 2);
  __bf16* v_l   = (__bf16*)alloc((size_t)TOK * D_ * 2);
  __bf16* yatt  = (__bf16*)alloc((size_t)TOK * D_ * 2);

  auto cvt = [&](const float* src, __bf16* dst, int n) {
    f32_to_bf16_kernel<<<(n + 255) / 256, 256, 0, stream>>>(src, dst, n);
  };
  cvt(x,    xb,    TOK * D_);
  cvt(Wabq, Wq_b,  NQ * D_);
  cvt(Wkv,  Wkv_b, NKV * D_);
  cvt(Wo,   Wo_b,  D_ * D_);

  // projections: abq = xb @ Wq^T ; abkv = xb @ Wkv^T
  gemm_bf16_nt<<<dim3((NQ + 127) / 128, TOK / 64), 256, 0, stream>>>(
      xb, Wq_b, abq, TOK, NQ, D_);
  gemm_bf16_nt<<<dim3((NKV + 127) / 128, TOK / 64), 256, 0, stream>>>(
      xb, Wkv_b, abkv, TOK, NKV, D_);

  // q/k/v build (RoPE + rank contraction)
  {
    int n = TOK * NHh * DHh;
    qkv_build_kernel<<<(n + 255) / 256, 256, 0, stream>>>(abq, abkv, q_l, k_l, v_l);
  }

  // causal flash attention
  attn_kernel<<<dim3(B_ * NHh, T_ / 16), 32, 0, stream>>>(q_l, k_l, v_l, yatt);

  // output projection: out = yatt @ Wo^T
  gemm_bf16_nt<<<dim3(D_ / 128, TOK / 64), 256, 0, stream>>>(
      yatt, Wo_b, out, TOK, D_, D_);
}
